// scNet_86535001080054
// MI455X (gfx1250) — matrix operand order repaired and checked by reading, not compiled, not run on hardware
//
#include <hip/hip_runtime.h>

// ---------------------------------------------------------------------------
// scNet pipeline for MI455X (gfx1250, wave32, WMMA).
// GEMMs run as bf16 WMMA (v_wmma_f32_16x16x32_bf16) with f32 accumulate,
// using a double-buffered LDS pipeline (global->regs overlapped with WMMA,
// one barrier per K-step). softmax / layernorm / l2norm / graph build / GCN
// scatter stay in f32.
// ---------------------------------------------------------------------------

#define NCELL  4096
#define HD     256
#define DHD    128
#define FFD    2048
#define KNB    6
#define NEDGE  (NCELL * KNB)

typedef __attribute__((ext_vector_type(16))) __bf16 v16bf;
typedef __attribute__((ext_vector_type(8)))  float  v8f;

union FragAB { v16bf v; uint4 u[2]; };
union Pack8  { uint4 q; __bf16 h[8]; };

#define CEIL(a, b) (((a) + (b) - 1) / (b))

// -------------------------------- elementwise ------------------------------

__global__ void k_cvt_bf16(__bf16* __restrict__ o, const float* __restrict__ x, int n) {
    int t = blockIdx.x * blockDim.x + threadIdx.x;
    if (t < n) o[t] = (__bf16)x[t];
}

__global__ void k_fill(float* p, float v, int n) {
    int t = blockIdx.x * blockDim.x + threadIdx.x;
    if (t < n) p[t] = v;
}

__global__ void k_relu_cvt(float* __restrict__ x, __bf16* __restrict__ xb, int n) {
    int t = blockIdx.x * blockDim.x + threadIdx.x;
    if (t < n) {
        float v = fmaxf(x[t], 0.0f);
        x[t]  = v;
        xb[t] = (__bf16)v;
    }
}

// ------------------------------ row l2norm (features) ----------------------

__global__ __launch_bounds__(64) void k_l2norm_rows(float* __restrict__ o,
                                                    const float* __restrict__ x, int D) {
    int i = blockIdx.x, t = threadIdx.x;
    __shared__ float red[64];
    float s = 0.0f;
    for (int d = t; d < D; d += 64) { float v = x[i * D + d]; s += v * v; }
    red[t] = s; __syncthreads();
    for (int st = 32; st > 0; st >>= 1) { if (t < st) red[t] += red[t + st]; __syncthreads(); }
    float inv = 1.0f / fmaxf(sqrtf(red[0]), 1e-12f);
    for (int d = t; d < D; d += 64) o[i * D + d] = x[i * D + d] * inv;
}

// ------------------------------ kNN graph build ----------------------------

// Feature graph: top-K by cosine similarity (rows pre-l2normed).
__global__ __launch_bounds__(256) void k_knn_feature(int* __restrict__ nbr,
                                                     float* __restrict__ w,
                                                     const float* __restrict__ f, int D) {
    int i = blockIdx.x, t = threadIdx.x;
    __shared__ float fi[64];
    __shared__ float ck[256 * KNB];
    __shared__ int   ci[256 * KNB];
    if (t < D) fi[t] = f[i * D + t];
    __syncthreads();

    float bk[KNB]; int bi[KNB];
    for (int r = 0; r < KNB; r++) { bk[r] = -1e30f; bi[r] = -1; }

    for (int j = t; j < NCELL; j += 256) {
        if (j == i) continue;
        float s = 0.0f;
        for (int d = 0; d < D; d++) s += fi[d] * f[j * D + d];
        if (s > bk[KNB - 1]) {
            int p = KNB - 1;
            while (p > 0 && s > bk[p - 1]) { bk[p] = bk[p - 1]; bi[p] = bi[p - 1]; p--; }
            bk[p] = s; bi[p] = j;
        }
    }
    for (int r = 0; r < KNB; r++) { ck[t * KNB + r] = bk[r]; ci[t * KNB + r] = bi[r]; }
    __syncthreads();
    if (t == 0) {
        for (int r = 0; r < KNB; r++) {
            float best = -1e30f; int bq = 0;
            for (int q = 0; q < 256 * KNB; q++)
                if (ck[q] > best) { best = ck[q]; bq = q; }
            nbr[i * KNB + r] = ci[bq];
            w[i * KNB + r]   = best;
            ck[bq] = -1e31f;
        }
    }
}

// Spatial graph: K smallest by (dist asc, sim desc); weight = 1/(dist+eps).
__global__ __launch_bounds__(256) void k_knn_spatial(int* __restrict__ nbr,
                                                     float* __restrict__ w,
                                                     const float* __restrict__ pos,
                                                     const float* __restrict__ f, int D) {
    int i = blockIdx.x, t = threadIdx.x;
    __shared__ float fi[64];
    __shared__ float pxy[2];
    __shared__ float cd[256 * KNB];
    __shared__ float cs[256 * KNB];
    __shared__ int   ci[256 * KNB];
    if (t < D) fi[t] = f[i * D + t];
    if (t < 2) pxy[t] = pos[i * 2 + t];
    __syncthreads();
    float px = pxy[0], py = pxy[1];
    float sqi = px * px + py * py;

    float bd[KNB], bs[KNB]; int bi[KNB];
    for (int r = 0; r < KNB; r++) { bd[r] = 1e30f; bs[r] = -1e30f; bi[r] = -1; }

    for (int j = t; j < NCELL; j += 256) {
        if (j == i) continue;
        float qx = pos[j * 2], qy = pos[j * 2 + 1];
        float d2 = sqi + qx * qx + qy * qy - 2.0f * (px * qx + py * qy);
        float dist = sqrtf(fmaxf(d2, 0.0f));
        float s = 0.0f;
        for (int d = 0; d < D; d++) s += fi[d] * f[j * D + d];
        bool better = (dist < bd[KNB - 1]) || (dist == bd[KNB - 1] && s > bs[KNB - 1]);
        if (better) {
            int p = KNB - 1;
            while (p > 0 && ((dist < bd[p - 1]) || (dist == bd[p - 1] && s > bs[p - 1]))) {
                bd[p] = bd[p - 1]; bs[p] = bs[p - 1]; bi[p] = bi[p - 1]; p--;
            }
            bd[p] = dist; bs[p] = s; bi[p] = j;
        }
    }
    for (int r = 0; r < KNB; r++) {
        cd[t * KNB + r] = bd[r]; cs[t * KNB + r] = bs[r]; ci[t * KNB + r] = bi[r];
    }
    __syncthreads();
    if (t == 0) {
        for (int r = 0; r < KNB; r++) {
            float bestd = 1e30f, bests = -1e30f; int bq = 0;
            for (int q = 0; q < 256 * KNB; q++) {
                if ((cd[q] < bestd) || (cd[q] == bestd && cs[q] > bests)) {
                    bestd = cd[q]; bests = cs[q]; bq = q;
                }
            }
            nbr[i * KNB + r] = ci[bq];
            w[i * KNB + r]   = 1.0f / (bestd + 1e-12f);
            cd[bq] = 1e31f;
        }
    }
}

// Single-block global min/max normalize of edge weights.
__global__ __launch_bounds__(256) void k_minmax_norm(float* __restrict__ w, int n) {
    __shared__ float smn[256], smx[256];
    int t = threadIdx.x;
    float mn = 1e30f, mx = -1e30f;
    for (int e = t; e < n; e += 256) { float v = w[e]; mn = fminf(mn, v); mx = fmaxf(mx, v); }
    smn[t] = mn; smx[t] = mx; __syncthreads();
    for (int st = 128; st > 0; st >>= 1) {
        if (t < st) { smn[t] = fminf(smn[t], smn[t + st]); smx[t] = fmaxf(smx[t], smx[t + st]); }
        __syncthreads();
    }
    mn = smn[0]; mx = smx[0];
    float inv = 1.0f / (mx - mn + 1e-12f);
    for (int e = t; e < n; e += 256) w[e] = (w[e] - mn) * inv;
}

// ------------------------------ GCN pieces ---------------------------------

__global__ void k_deg_acc(float* __restrict__ deg, const int* __restrict__ dst,
                          const float* __restrict__ w, int n) {
    int t = blockIdx.x * blockDim.x + threadIdx.x;
    if (t < n) atomicAdd(&deg[dst[t]], w[t]);
}

__global__ void k_dinv(float* __restrict__ dinv, const float* __restrict__ deg, int n) {
    int t = blockIdx.x * blockDim.x + threadIdx.x;
    if (t < n) dinv[t] = deg[t] > 0.0f ? rsqrtf(deg[t]) : 0.0f;
}

// Small-K f32 GEMM: C[M x Np] = A[M x Kd] @ B[Kd x Np]  (GCN projection)
__global__ __launch_bounds__(256) void k_gemm_f32(float* __restrict__ C,
                                                  const float* __restrict__ A,
                                                  const float* __restrict__ B,
                                                  int M, int Np, int Kd) {
    int t = blockIdx.x * 256 + threadIdx.x;
    if (t >= M * Np) return;
    int m = t / Np, n = t % Np;
    float s = 0.0f;
    for (int k = 0; k < Kd; k++) s += A[m * Kd + k] * B[k * Np + n];
    C[t] = s;
}

__global__ void k_gcn_self(float* __restrict__ h, const float* __restrict__ xw,
                           const float* __restrict__ dinv, const float* __restrict__ b) {
    int i = blockIdx.x, c = threadIdx.x;
    float di = dinv[i];
    h[i * HD + c] = b[c] + di * di * xw[i * HD + c];
}

__global__ void k_gcn_scatter(float* __restrict__ h, const float* __restrict__ xw,
                              const float* __restrict__ dinv, const int* __restrict__ dst,
                              const float* __restrict__ w) {
    int e = blockIdx.x, c = threadIdx.x;
    int s = e / KNB, d = dst[e];
    float nm = dinv[s] * w[e] * dinv[d];
    atomicAdd(&h[d * HD + c], nm * xw[s * HD + c]);
}

// ------------------------------ WMMA GEMM ----------------------------------
// C[M x Np](f32) = scale * A[M x Kd](bf16) * B(bf16) + bias, optional ReLU.
// bT=1: B is [Np x Kd] row-major (i.e. C = A * B^T, K contiguous per row).
// bT=0: B is [Kd x Np] row-major (C = A * B), staged transposed via LDS.
// Block tile 128x128, 8 wave32 waves, wave tile 64x32 (4x2 WMMA subtiles).
// Double-buffered LDS pipeline: next tile's global loads issued before the
// WMMA burst on the current buffer; one barrier per K-step.
// Requires M,Np multiples of 128 and Kd multiple of 32 (true for all calls).

__global__ __launch_bounds__(256)
void k_gemm_bf16(float* __restrict__ C, const __bf16* __restrict__ A,
                 const __bf16* __restrict__ B, const float* __restrict__ bias,
                 int M, int Np, int Kd, int lda, int ldb, int ldc,
                 float scale, int relu, int bT) {
    __shared__ __align__(16) __bf16 As[2][128][32];
    __shared__ __align__(16) __bf16 Bs[2][128][32];

    const int t    = threadIdx.x;
    const int lane = t & 31;
    const int wv   = t >> 5;
    const int l16  = lane & 15;
    const int lh   = lane >> 4;            // 0 or 1 (half-wave)
    const int wm   = (wv & 1) * 64;        // wave row offset in tile
    const int wn   = (wv >> 1) * 32;       // wave col offset in tile
    const int rowBase = blockIdx.y * 128;
    const int colBase = blockIdx.x * 128;

    const v8f vz = {0.f, 0.f, 0.f, 0.f, 0.f, 0.f, 0.f, 0.f};
    v8f acc[4][2];
    for (int i = 0; i < 4; i++)
        for (int j = 0; j < 2; j++) acc[i][j] = vz;

    const int arow = t >> 1, aseg = (t & 1) * 16;   // A/B(T) tile loader: 128 x 32
    const int krow = t >> 3, nseg = (t & 7) * 16;   // bKN loader: 32 x 128

    // ---- global load of this thread's portion of tile at K offset kk ----
    auto load_tile = [&](int kk, uint4& a0, uint4& a1, uint4& b0, uint4& b1) {
        const uint4* sa = reinterpret_cast<const uint4*>(
            &A[(size_t)(rowBase + arow) * lda + kk + aseg]);
        a0 = sa[0]; a1 = sa[1];
        if (bT) {
            const uint4* sb = reinterpret_cast<const uint4*>(
                &B[(size_t)(colBase + arow) * ldb + kk + aseg]);
            b0 = sb[0]; b1 = sb[1];
        } else {
            const uint4* sb = reinterpret_cast<const uint4*>(
                &B[(size_t)(kk + krow) * ldb + colBase + nseg]);
            b0 = sb[0]; b1 = sb[1];
        }
    };
    // ---- store staged registers into LDS buffer `buf` ----
    auto store_tile = [&](int buf, uint4 a0, uint4 a1, uint4 b0, uint4 b1) {
        uint4* da = reinterpret_cast<uint4*>(&As[buf][arow][aseg]);
        da[0] = a0; da[1] = a1;
        if (bT) {
            uint4* db = reinterpret_cast<uint4*>(&Bs[buf][arow][aseg]);
            db[0] = b0; db[1] = b1;
        } else {
            Pack8 p0, p1;
            p0.q = b0; p1.q = b1;
            #pragma unroll
            for (int e = 0; e < 8; e++) {
                Bs[buf][nseg + e][krow]     = p0.h[e];
                Bs[buf][nseg + 8 + e][krow] = p1.h[e];
            }
        }
    };

    const int nsteps = Kd >> 5;
    uint4 a0, a1, b0, b1;

    // prologue: stage tile 0
    load_tile(0, a0, a1, b0, b1);
    store_tile(0, a0, a1, b0, b1);
    __syncthreads();

    for (int s = 0; s < nsteps; s++) {
        const int cur = s & 1;
        const bool more = (s + 1) < nsteps;

        // issue next tile's global loads; latency hidden under the WMMA burst
        if (more) load_tile((s + 1) << 5, a0, a1, b0, b1);
        if (s + 2 < nsteps)
            __builtin_prefetch(&A[(size_t)(rowBase + arow) * lda + ((s + 2) << 5) + aseg], 0, 1);

        // fragment loads per documented CDNA5 bf16 WMMA VGPR layouts
        FragAB af[4], bf[2];
        #pragma unroll
        for (int i = 0; i < 4; i++) {
            const uint4* pr = reinterpret_cast<const uint4*>(&As[cur][wm + i * 16 + l16][0]);
            af[i].u[0] = pr[lh];         // K = lh*8 .. +7
            af[i].u[1] = pr[lh + 2];     // K = lh*8+16 .. +7
        }
        #pragma unroll
        for (int j = 0; j < 2; j++) {
            const uint4* pc = reinterpret_cast<const uint4*>(&Bs[cur][wn + j * 16 + l16][0]);
            bf[j].u[0] = pc[lh];         // K = lh*8 .. +7
            bf[j].u[1] = pc[lh + 2];     // K = lh*8+16 .. +7
        }
        #pragma unroll
        for (int i = 0; i < 4; i++)
            #pragma unroll
            for (int j = 0; j < 2; j++)
                acc[i][j] = __builtin_amdgcn_wmma_f32_16x16x32_bf16(
                    false, af[i].v, false, bf[j].v, (short)0, acc[i][j], false, false);

        // stage next tile into the other buffer, then one barrier per step
        if (more) store_tile(cur ^ 1, a0, a1, b0, b1);
        __syncthreads();
    }

    // epilogue: C layout VGPR r -> M = r + lh*8, N = l16
    #pragma unroll
    for (int j = 0; j < 2; j++) {
        int col = colBase + wn + j * 16 + l16;
        float bs = bias ? bias[col] : 0.0f;
        #pragma unroll
        for (int i = 0; i < 4; i++) {
            #pragma unroll
            for (int r = 0; r < 8; r++) {
                int row = rowBase + wm + i * 16 + lh * 8 + r;
                float v = acc[i][j][r] * scale + bs;
                if (relu) v = fmaxf(v, 0.0f);
                C[(size_t)row * ldc + col] = v;
            }
        }
    }
}

// ------------------------------ softmax / LN / l2norm ----------------------

__global__ __launch_bounds__(256) void k_softmax_bf16(__bf16* __restrict__ o,
                                                      const float* __restrict__ in) {
    int i = blockIdx.x, t = threadIdx.x;
    __shared__ float red[256];
    const float* row = in + (size_t)i * NCELL;
    float lv[16];
    float mx = -1e30f;
    #pragma unroll
    for (int u = 0; u < 16; u++) { lv[u] = row[u * 256 + t]; mx = fmaxf(mx, lv[u]); }
    red[t] = mx; __syncthreads();
    for (int st = 128; st > 0; st >>= 1) { if (t < st) red[t] = fmaxf(red[t], red[t + st]); __syncthreads(); }
    float m = red[0]; __syncthreads();
    float s = 0.0f;
    #pragma unroll
    for (int u = 0; u < 16; u++) { lv[u] = __expf(lv[u] - m); s += lv[u]; }
    red[t] = s; __syncthreads();
    for (int st = 128; st > 0; st >>= 1) { if (t < st) red[t] += red[t + st]; __syncthreads(); }
    float inv = 1.0f / red[0];
    #pragma unroll
    for (int u = 0; u < 16; u++)
        o[(size_t)i * NCELL + u * 256 + t] = (__bf16)(lv[u] * inv);
}

__global__ __launch_bounds__(256) void k_add_ln(float* __restrict__ of, __bf16* __restrict__ ob,
                                                const float* __restrict__ a, const float* __restrict__ b,
                                                const float* __restrict__ g, const float* __restrict__ bt) {
    int i = blockIdx.x, t = threadIdx.x;
    __shared__ float r1[256], r2[256];
    float v = a[i * HD + t] + (b ? b[i * HD + t] : 0.0f);
    r1[t] = v; r2[t] = v * v; __syncthreads();
    for (int st = 128; st > 0; st >>= 1) {
        if (t < st) { r1[t] += r1[t + st]; r2[t] += r2[t + st]; }
        __syncthreads();
    }
    float mu  = r1[0] * (1.0f / HD);
    float var = r2[0] * (1.0f / HD) - mu * mu;
    float o = (v - mu) * rsqrtf(var + 1e-5f) * g[t] + bt[t];
    of[i * HD + t] = o;
    if (ob) ob[i * HD + t] = (__bf16)o;
}

__global__ __launch_bounds__(256) void k_l2norm_embed(float* __restrict__ of,
                                                      __bf16* __restrict__ ob,
                                                      const float* __restrict__ x) {
    int i = blockIdx.x, t = threadIdx.x;
    __shared__ float red[256];
    float v = x[i * HD + t];
    red[t] = v * v; __syncthreads();
    for (int st = 128; st > 0; st >>= 1) { if (t < st) red[t] += red[t + st]; __syncthreads(); }
    float inv = 1.0f / fmaxf(sqrtf(red[0]), 1e-12f);
    of[i * HD + t] = v * inv;
    ob[i * HD + t] = (__bf16)(v * inv);
}

// ------------------------------ host orchestration -------------------------

struct Enc {
    const float *gcn_w, *gcn_b, *in_b, *out_b, *ln1_g, *ln1_b, *ff1_b, *ff2_b, *ln2_g, *ln2_b;
    const __bf16 *in_w, *out_w, *ff1_w, *ff2_w;
    int D;
};

struct Scratch {
    float *deg, *dinv, *xw, *x, *qkv, *scores, *o, *attn, *y, *ff, *g2, *z;
    __bf16 *xb, *qkvb, *ab, *ob, *yb, *ffb;
};

static void launch_gemm(hipStream_t st, float* C, const __bf16* A, const __bf16* B,
                        const float* bias, int M, int Np, int Kd,
                        int lda, int ldb, int ldc, float scale, int relu, int bT) {
    dim3 g(Np / 128, M / 128);
    k_gemm_bf16<<<g, 256, 0, st>>>(C, A, B, bias, M, Np, Kd, lda, ldb, ldc, scale, relu, bT);
}

static void run_encoder(hipStream_t st, const float* feat, const int* edst, const float* ew,
                        const Enc& P, float* emb_f32, __bf16* emb_b16, const Scratch& S) {
    const float inv_sqrt_dh = 1.0f / sqrtf((float)DHD);

    // degree norm
    k_fill<<<CEIL(NCELL, 256), 256, 0, st>>>(S.deg, 1.0f, NCELL);              // self-loop weight
    k_deg_acc<<<CEIL(NEDGE, 256), 256, 0, st>>>(S.deg, edst, ew, NEDGE);
    k_dinv<<<CEIL(NCELL, 256), 256, 0, st>>>(S.dinv, S.deg, NCELL);

    // xw = feat @ gcn_w ; h = self + scatter ; x = relu(h)
    k_gemm_f32<<<CEIL(NCELL * HD, 256), 256, 0, st>>>(S.xw, feat, P.gcn_w, NCELL, HD, P.D);
    k_gcn_self<<<NCELL, HD, 0, st>>>(S.x, S.xw, S.dinv, P.gcn_b);
    k_gcn_scatter<<<NEDGE, HD, 0, st>>>(S.x, S.xw, S.dinv, edst, ew);
    k_relu_cvt<<<CEIL(NCELL * HD, 256), 256, 0, st>>>(S.x, S.xb, NCELL * HD);

    // qkv
    launch_gemm(st, S.qkv, S.xb, P.in_w, P.in_b, NCELL, 3 * HD, HD, HD, HD, 3 * HD, 1.0f, 0, 1);
    k_cvt_bf16<<<CEIL(NCELL * 3 * HD, 256), 256, 0, st>>>(S.qkvb, S.qkv, NCELL * 3 * HD);

    // attention per head
    for (int h = 0; h < 2; h++) {
        launch_gemm(st, S.scores, S.qkvb + h * DHD, S.qkvb + HD + h * DHD, nullptr,
                    NCELL, NCELL, DHD, 3 * HD, 3 * HD, NCELL, inv_sqrt_dh, 0, 1);
        k_softmax_bf16<<<NCELL, 256, 0, st>>>(S.ab, S.scores);
        launch_gemm(st, S.o + h * DHD, S.ab, S.qkvb + 2 * HD + h * DHD, nullptr,
                    NCELL, DHD, NCELL, NCELL, 3 * HD, HD, 1.0f, 0, 0);
    }
    k_cvt_bf16<<<CEIL(NCELL * HD, 256), 256, 0, st>>>(S.ob, S.o, NCELL * HD);

    // out projection, residual + LN1
    launch_gemm(st, S.attn, S.ob, P.out_w, P.out_b, NCELL, HD, HD, HD, HD, HD, 1.0f, 0, 1);
    k_add_ln<<<NCELL, HD, 0, st>>>(S.y, S.yb, S.x, S.attn, P.ln1_g, P.ln1_b);

    // FFN, residual + LN2
    launch_gemm(st, S.ff, S.yb, P.ff1_w, P.ff1_b, NCELL, FFD, HD, HD, HD, FFD, 1.0f, 1, 1);
    k_cvt_bf16<<<CEIL(NCELL * FFD, 256), 256, 0, st>>>(S.ffb, S.ff, NCELL * FFD);
    launch_gemm(st, S.g2, S.ffb, P.ff2_w, P.ff2_b, NCELL, HD, FFD, FFD, FFD, HD, 1.0f, 0, 1);
    k_add_ln<<<NCELL, HD, 0, st>>>(S.z, nullptr, S.y, S.g2, P.ln2_g, P.ln2_b);

    // l2-normalized embedding
    k_l2norm_embed<<<NCELL, HD, 0, st>>>(emb_f32, emb_b16, S.z);
}

extern "C" void kernel_launch(void* const* d_in, const int* in_sizes, int n_in,
                              void* d_out, int out_size, void* d_ws, size_t ws_size,
                              hipStream_t stream) {
    (void)in_sizes; (void)n_in; (void)out_size; (void)ws_size;

    const float* m1  = (const float*)d_in[0];   // modality1 [N x 50]
    const float* m2  = (const float*)d_in[1];   // modality2 [N x 30]
    const float* pos = (const float*)d_in[2];   // position  [N x 2]

    // params flattened dict order: atac (14 leaves) then rna (14 leaves)
    auto F = [&](int i) { return (const float*)d_in[i]; };
    const int A0 = 3, R0 = 17;

    // ---- workspace layout ----
    char* w = (char*)d_ws;
    size_t off = 0;
    auto alloc = [&](size_t bytes) -> void* {
        void* p = w + off;
        off = (off + bytes + 255) & ~(size_t)255;
        return p;
    };

    __bf16* wbf[8]; // bf16 weights: [enc][in_w,out_w,ff1_w,ff2_w]
    const int wsz[4] = {3 * HD * HD, HD * HD, FFD * HD, HD * FFD};
    for (int e = 0; e < 2; e++)
        for (int q = 0; q < 4; q++)
            wbf[e * 4 + q] = (__bf16*)alloc((size_t)wsz[q] * 2);

    float* fn1 = (float*)alloc((size_t)NCELL * 50 * 4);
    float* fn2 = (float*)alloc((size_t)NCELL * 30 * 4);

    int*   gdst[4]; float* gwt[4];
    for (int g = 0; g < 4; g++) {
        gdst[g] = (int*)alloc((size_t)NEDGE * 4);
        gwt[g]  = (float*)alloc((size_t)NEDGE * 4);
    }

    Scratch S;
    S.deg    = (float*)alloc((size_t)NCELL * 4);
    S.dinv   = (float*)alloc((size_t)NCELL * 4);
    S.xw     = (float*)alloc((size_t)NCELL * HD * 4);
    S.x      = (float*)alloc((size_t)NCELL * HD * 4);
    S.xb     = (__bf16*)alloc((size_t)NCELL * HD * 2);
    S.qkv    = (float*)alloc((size_t)NCELL * 3 * HD * 4);
    S.qkvb   = (__bf16*)alloc((size_t)NCELL * 3 * HD * 2);
    S.scores = (float*)alloc((size_t)NCELL * NCELL * 4);
    S.ab     = (__bf16*)alloc((size_t)NCELL * NCELL * 2);
    S.o      = (float*)alloc((size_t)NCELL * HD * 4);
    S.ob     = (__bf16*)alloc((size_t)NCELL * HD * 2);
    S.attn   = (float*)alloc((size_t)NCELL * HD * 4);
    S.y      = (float*)alloc((size_t)NCELL * HD * 4);
    S.yb     = (__bf16*)alloc((size_t)NCELL * HD * 2);
    S.ff     = (float*)alloc((size_t)NCELL * FFD * 4);
    S.ffb    = (__bf16*)alloc((size_t)NCELL * FFD * 2);
    S.g2     = (float*)alloc((size_t)NCELL * HD * 4);
    S.z      = (float*)alloc((size_t)NCELL * HD * 4);

    __bf16* emb_b16[4];
    for (int e = 0; e < 4; e++) emb_b16[e] = (__bf16*)alloc((size_t)NCELL * HD * 2);

    // ---- convert weights to bf16 ----
    const int widx[4] = {2, 4, 8, 10}; // in_w, out_w, ff1_w, ff2_w offsets in enc param block
    for (int e = 0; e < 2; e++) {
        int base = e == 0 ? A0 : R0;
        for (int q = 0; q < 4; q++)
            k_cvt_bf16<<<CEIL(wsz[q], 256), 256, 0, stream>>>(wbf[e * 4 + q], F(base + widx[q]), wsz[q]);
    }

    // ---- l2norm features ----
    k_l2norm_rows<<<NCELL, 64, 0, stream>>>(fn1, m1, 50);
    k_l2norm_rows<<<NCELL, 64, 0, stream>>>(fn2, m2, 30);

    // ---- build graphs: 0=spatial(m2/atac), 1=spatial(m1/rna), 2=feat(m2), 3=feat(m1)
    k_knn_spatial<<<NCELL, 256, 0, stream>>>(gdst[0], gwt[0], pos, fn2, 30);
    k_knn_spatial<<<NCELL, 256, 0, stream>>>(gdst[1], gwt[1], pos, fn1, 50);
    k_knn_feature<<<NCELL, 256, 0, stream>>>(gdst[2], gwt[2], fn2, 30);
    k_knn_feature<<<NCELL, 256, 0, stream>>>(gdst[3], gwt[3], fn1, 50);
    for (int g = 0; g < 4; g++)
        k_minmax_norm<<<1, 256, 0, stream>>>(gwt[g], NEDGE);

    // ---- encoder params ----
    Enc enc[2];
    for (int e = 0; e < 2; e++) {
        int b = e == 0 ? A0 : R0;
        enc[e].gcn_w = F(b + 0);  enc[e].gcn_b = F(b + 1);
        enc[e].in_b  = F(b + 3);  enc[e].out_b = F(b + 5);
        enc[e].ln1_g = F(b + 6);  enc[e].ln1_b = F(b + 7);
        enc[e].ff1_b = F(b + 9);  enc[e].ff2_b = F(b + 11);
        enc[e].ln2_g = F(b + 12); enc[e].ln2_b = F(b + 13);
        enc[e].in_w  = wbf[e * 4 + 0];
        enc[e].out_w = wbf[e * 4 + 1];
        enc[e].ff1_w = wbf[e * 4 + 2];
        enc[e].ff2_w = wbf[e * 4 + 3];
        enc[e].D     = e == 0 ? 30 : 50;
    }

    // ---- outputs: (ea, er, eaf, erf, sim) ----
    float* out = (float*)d_out;
    float* ea  = out;
    float* er  = out + (size_t)NCELL * HD;
    float* eaf = out + 2 * (size_t)NCELL * HD;
    float* erf = out + 3 * (size_t)NCELL * HD;
    float* sim = out + 4 * (size_t)NCELL * HD;

    run_encoder(stream, m2, gdst[0], gwt[0], enc[0], ea,  emb_b16[0], S);
    run_encoder(stream, m1, gdst[1], gwt[1], enc[1], er,  emb_b16[1], S);
    run_encoder(stream, m2, gdst[2], gwt[2], enc[0], eaf, emb_b16[2], S);
    run_encoder(stream, m1, gdst[3], gwt[3], enc[1], erf, emb_b16[3], S);

    // sim = er @ ea^T
    launch_gemm(stream, sim, emb_b16[1], emb_b16[0], nullptr,
                NCELL, NCELL, HD, HD, HD, NCELL, 1.0f, 0, 1);
}
